// FDM_90297392431829
// MI455X (gfx1250) — matrix operand order repaired and checked by reading, not compiled, not run on hardware
//
#include <hip/hip_runtime.h>
#include <hip/hip_bf16.h>
#include <math.h>

typedef __attribute__((ext_vector_type(16))) _Float16 v16h;
typedef __attribute__((ext_vector_type(8)))  _Float16 v8h;
typedef __attribute__((ext_vector_type(8)))  float    v8f;

#define BATCH 2
#define CIN   256
#define PL    128
#define HW    4096
#define HWDIM 64

// ---------------------------------------------------------------------------
// WMMA fragment loaders for CDNA5 V_WMMA_F32_16X16X32_F16 (wave32).
// A (16x32, MxK), 16-bit: lanes 0-15 = rows M, VGPR0-3 hold K=0..7 (half 0)
// or K=8..15 (half 1); VGPR4-7 hold K=16..23 / 24..31.
// B (32x16, KxN), 16-bit: lane%16 = column N, same K layout as A:
// K=8*half..+7 then K=16+8*half..+7.
// C/D (16x16 f32): lane%16 = N, VGPR v = row (v + 8*half).
// ---------------------------------------------------------------------------

// A fragment from row-major f16 src: element [m][k] at src[m*ld + k]
__device__ __forceinline__ v16h load_a_f16(const _Float16* src, int ld, int m0, int k0) {
    int lane = threadIdx.x & 31;
    int mr = lane & 15, half = lane >> 4;
    const _Float16* row = src + (size_t)(m0 + mr) * ld + k0 + 8 * half;
    v16h a;
#pragma unroll
    for (int e = 0; e < 8; ++e) a[e] = row[e];
#pragma unroll
    for (int e = 8; e < 16; ++e) a[e] = row[e + 8];
    return a;
}

// A fragment from row-major f32 src (convert to f16 on load)
__device__ __forceinline__ v16h load_a_f32(const float* src, int ld, int m0, int k0) {
    int lane = threadIdx.x & 31;
    int mr = lane & 15, half = lane >> 4;
    const float* row = src + (size_t)(m0 + mr) * ld + k0 + 8 * half;
    v16h a;
#pragma unroll
    for (int e = 0; e < 8; ++e) a[e] = (_Float16)row[e];
#pragma unroll
    for (int e = 8; e < 16; ++e) a[e] = (_Float16)row[e + 8];
    return a;
}

// B fragment where element [k][n] lives at src[n*ld + k] (i.e. B stored
// "column-major" / N-major). Per-lane reads are two 8-half (16B) runs at offsets 0 and 16.
__device__ __forceinline__ v16h load_b_t(const _Float16* src, int ld, int k0, int n0) {
    int lane = threadIdx.x & 31;
    int mr = lane & 15, half = lane >> 4;
    const _Float16* col = src + (size_t)(n0 + mr) * ld + k0 + 8 * half;
    v16h b;
#pragma unroll
    for (int e = 0; e < 8; ++e) b[e] = col[e];
#pragma unroll
    for (int e = 8; e < 16; ++e) b[e] = col[e + 8];
    return b;
}

// ---------------------------------------------------------------------------
// K0: x [b][c][q] f32 -> xT [b][q][c] f16 ; click -> wclick = click*0.9+0.1
// ---------------------------------------------------------------------------
__global__ void cvt_kernel(const float* __restrict__ x, _Float16* __restrict__ xT,
                           const float* __restrict__ click, float* __restrict__ wclick) {
    int idx = blockIdx.x * blockDim.x + threadIdx.x;
    if (idx < BATCH * HW * CIN) {
        int c = idx & (CIN - 1);
        int q = (idx >> 8) & (HW - 1);
        int b = idx >> 20;
        xT[idx] = (_Float16)x[((size_t)(b * CIN + c)) * HW + q];
    }
    if (idx < BATCH * 2 * HW) {
        wclick[idx] = click[idx] * 0.9f + 0.1f;
    }
}

// ---------------------------------------------------------------------------
// K1: projections t/p/g = W * x.  M=3*128, K=256, N=4096 per batch.
// t,p stored transposed [q][128] f16 (q-major); g stored [128][4096] (c-major).
// One wave computes one 16x16 tile (8 WMMAs over K=256).
// ---------------------------------------------------------------------------
__global__ __launch_bounds__(32) void proj_kernel(
    const float* __restrict__ w_t, const float* __restrict__ w_p,
    const float* __restrict__ w_g, const _Float16* __restrict__ xT,
    _Float16* __restrict__ tT, _Float16* __restrict__ pT, _Float16* __restrict__ gC) {
    int b = blockIdx.z;
    int mt = blockIdx.y;           // 0..23  (sel = mt/8, 128 rows each)
    int q0 = blockIdx.x * 16;
    int sel = mt >> 3;
    int rm0 = (mt & 7) * 16;
    const float* W = (sel == 0) ? w_t : ((sel == 1) ? w_p : w_g);
    const _Float16* xb_ = xT + (size_t)b * HW * CIN;

    v8f acc = {};
#pragma unroll
    for (int kc = 0; kc < 8; ++kc) {
        v16h a = load_a_f32(W, CIN, rm0, kc * 32);
        v16h bb = load_b_t(xb_, CIN, kc * 32, q0);
        acc = __builtin_amdgcn_wmma_f32_16x16x32_f16(false, a, false, bb,
                                                     (short)0, acc, false, false);
    }

    int lane = threadIdx.x & 31, mr = lane & 15, half = lane >> 4;
    if (sel < 2) {
        // transposed store: [q][ch], 8 consecutive f16 per lane (16B aligned)
        _Float16* dst = ((sel == 0) ? tT : pT) +
                        ((size_t)b * HW + q0 + mr) * PL + rm0 + 8 * half;
        v8h outv;
#pragma unroll
        for (int v = 0; v < 8; ++v) outv[v] = (_Float16)acc[v];
        *(v8h*)dst = outv;
    } else {
        // g stored channel-major [128][4096]
        _Float16* dst = gC + (size_t)b * PL * HW;
#pragma unroll
        for (int v = 0; v < 8; ++v)
            dst[(size_t)(rm0 + v + 8 * half) * HW + q0 + mr] = (_Float16)acc[v];
    }
}

// ---------------------------------------------------------------------------
// K2: gate head. One block (256 thr) per pixel: depthwise 3x3 -> pointwise
// 256x256 + ReLU -> 1x256 head -> sigmoid. Writes gate_pos/neg_ori to d_out
// and scaled gates (0.9*g+0.1) to workspace.
// ---------------------------------------------------------------------------
__global__ __launch_bounds__(256) void gate_kernel(
    const float* __restrict__ x,
    const float* __restrict__ dw_w, const float* __restrict__ dw_b,
    const float* __restrict__ pw_w, const float* __restrict__ pw_b,
    const float* __restrict__ head_w, const float* __restrict__ head_b,
    float* __restrict__ gate_pos_out, float* __restrict__ gate_neg_out,
    float* __restrict__ gpos_s, float* __restrict__ gneg_s) {
    int pix = blockIdx.x;                  // b*4096 + q
    int b = pix >> 12, q = pix & (HW - 1);
    int py = q >> 6, px = q & 63;
    int c = threadIdx.x;

    __shared__ float y1[CIN];
    __shared__ float red[CIN];

    const float* xc = x + ((size_t)(b * CIN + c)) * HW;
    float acc = dw_b[c];
#pragma unroll
    for (int dy = -1; dy <= 1; ++dy)
#pragma unroll
        for (int dx = -1; dx <= 1; ++dx) {
            int yy = py + dy, xx = px + dx;
            if (yy >= 0 && yy < HWDIM && xx >= 0 && xx < HWDIM)
                acc += dw_w[c * 9 + (dy + 1) * 3 + (dx + 1)] * xc[yy * HWDIM + xx];
        }
    y1[c] = acc;
    __syncthreads();

    float a2 = pw_b[c];
    const float* wrow = pw_w + (size_t)c * CIN;
    for (int k = 0; k < CIN; ++k) a2 += wrow[k] * y1[k];
    a2 = fmaxf(a2, 0.0f);
    red[c] = head_w[c] * a2;
    __syncthreads();
    for (int s = 128; s > 0; s >>= 1) {
        if (c < s) red[c] += red[c + s];
        __syncthreads();
    }
    if (c == 0) {
        float gp = 1.0f / (1.0f + __expf(-(red[0] + head_b[0])));
        int o = b * HW + q;
        gate_pos_out[o] = gp;
        gate_neg_out[o] = 1.0f - gp;
        gpos_s[o] = gp * 0.9f + 0.1f;
        gneg_s[o] = (1.0f - gp) * 0.9f + 0.1f;
    }
}

// ---------------------------------------------------------------------------
// K4: fused attention per branch, flash-style. One wave = one 16-query tile.
// Streams 32-key tiles: S = t_q^T p_k (8 WMMAs), key-weight, online softmax,
// P routed through LDS to A-layout, xb += P * g (8 WMMAs). Epilogue applies
// gate[q]/l[q] and stores the 16x128 xb tile (f16, q-major).
// ---------------------------------------------------------------------------
__global__ __launch_bounds__(32) void attn_kernel(
    const _Float16* __restrict__ tT, const _Float16* __restrict__ pT,
    const _Float16* __restrict__ gC, const float* __restrict__ wclick,
    const float* __restrict__ gpos_s, const float* __restrict__ gneg_s,
    _Float16* __restrict__ xbT) {
    int b = blockIdx.z, s = blockIdx.y;
    int q0 = blockIdx.x * 16;
    int lane = threadIdx.x & 31, mr = lane & 15, half = lane >> 4;

    const _Float16* tb = tT + (size_t)b * HW * PL;
    const _Float16* pb = pT + (size_t)b * HW * PL;
    const _Float16* gb = gC + (size_t)b * PL * HW;
    const float* wm = wclick + (size_t)(b * 2 + s) * HW;
    const float* gate = ((s == 0) ? gpos_s : gneg_s) + (size_t)b * HW;

    // query A-fragments (invariant over the key loop): t^T, 16q x 128c
    v16h at[4];
#pragma unroll
    for (int kc = 0; kc < 4; ++kc) at[kc] = load_a_f16(tb, PL, q0, kc * 32);

    v8f zero = {};
    v8f acc[8];
#pragma unroll
    for (int cc = 0; cc < 8; ++cc) acc[cc] = zero;
    float mrow[8], lrow[8];
#pragma unroll
    for (int v = 0; v < 8; ++v) { mrow[v] = -1e30f; lrow[v] = 0.0f; }

    __shared__ _Float16 lP[16 * 32];
    __shared__ _Float16 lxb[16 * PL];

    for (int kt = 0; kt < HW / 32; ++kt) {
        int k0 = kt * 32;
        // ---- scores for 16q x 32k (two 16x16 C tiles), K = 128 channels ----
        v8f S0 = zero, S1 = zero;
#pragma unroll
        for (int kc = 0; kc < 4; ++kc) {
            v16h b0 = load_b_t(pb, PL, kc * 32, k0);
            v16h b1 = load_b_t(pb, PL, kc * 32, k0 + 16);
            S0 = __builtin_amdgcn_wmma_f32_16x16x32_f16(false, at[kc], false, b0,
                                                        (short)0, S0, false, false);
            S1 = __builtin_amdgcn_wmma_f32_16x16x32_f16(false, at[kc], false, b1,
                                                        (short)0, S1, false, false);
        }
        // key-side click weights (per key column)
        float w0 = wm[k0 + mr], w1 = wm[k0 + 16 + mr];
#pragma unroll
        for (int v = 0; v < 8; ++v) { S0[v] *= w0; S1[v] *= w1; }

        // ---- online softmax: row stats via 16-lane xor reductions ----
        float tm[8];
#pragma unroll
        for (int v = 0; v < 8; ++v) tm[v] = fmaxf(S0[v], S1[v]);
#pragma unroll
        for (int mask = 1; mask <= 8; mask <<= 1)
#pragma unroll
            for (int v = 0; v < 8; ++v) tm[v] = fmaxf(tm[v], __shfl_xor(tm[v], mask, 32));

        float scl[8];
#pragma unroll
        for (int v = 0; v < 8; ++v) {
            float nm = fmaxf(mrow[v], tm[v]);
            scl[v] = __expf(mrow[v] - nm);
            mrow[v] = nm;
            S0[v] = __expf(S0[v] - nm);
            S1[v] = __expf(S1[v] - nm);
        }
        float rs[8];
#pragma unroll
        for (int v = 0; v < 8; ++v) rs[v] = S0[v] + S1[v];
#pragma unroll
        for (int mask = 1; mask <= 8; mask <<= 1)
#pragma unroll
            for (int v = 0; v < 8; ++v) rs[v] += __shfl_xor(rs[v], mask, 32);
#pragma unroll
        for (int v = 0; v < 8; ++v) lrow[v] = lrow[v] * scl[v] + rs[v];
        // rescale running accumulators (row layout matches C fragments)
#pragma unroll
        for (int cc = 0; cc < 8; ++cc)
#pragma unroll
            for (int v = 0; v < 8; ++v) acc[cc][v] *= scl[v];

        // ---- re-layout P: C fragment -> LDS 16x32 -> A fragment ----
#pragma unroll
        for (int v = 0; v < 8; ++v) {
            int row = v + 8 * half;
            lP[row * 32 + mr] = (_Float16)S0[v];
            lP[row * 32 + 16 + mr] = (_Float16)S1[v];
        }
        __syncthreads();
        v16h ap;
        {
            int base = mr * 32 + 8 * half;
#pragma unroll
            for (int e = 0; e < 8; ++e) ap[e] = lP[base + e];
#pragma unroll
            for (int e = 8; e < 16; ++e) ap[e] = lP[base + e + 8];
        }
        __syncthreads();

        // ---- xb(16q x 128c) += P(16x32) * g(32k x 128c) ----
#pragma unroll
        for (int cc = 0; cc < 8; ++cc) {
            v16h bg = load_b_t(gb, HW, k0, cc * 16);  // g is [c][k] c-major
            acc[cc] = __builtin_amdgcn_wmma_f32_16x16x32_f16(false, ap, false, bg,
                                                             (short)0, acc[cc], false, false);
        }
    }

    // ---- epilogue: query gate / softmax denominator, store f16 tile ----
    float fac[8];
#pragma unroll
    for (int v = 0; v < 8; ++v) {
        int row = v + 8 * half;
        fac[v] = gate[q0 + row] / lrow[v];
    }
#pragma unroll
    for (int cc = 0; cc < 8; ++cc)
#pragma unroll
        for (int v = 0; v < 8; ++v)
            lxb[(v + 8 * half) * PL + cc * 16 + mr] = (_Float16)(acc[cc][v] * fac[v]);
    __syncthreads();
    {
        int r = lane >> 1, hh = (lane & 1) * 64;
        _Float16* dst = xbT + ((size_t)(b * 2 + s) * HW + q0 + r) * PL + hh;
        const _Float16* srcl = lxb + r * PL + hh;
#pragma unroll
        for (int t = 0; t < 8; ++t) *(v8h*)(dst + t * 8) = *(const v8h*)(srcl + t * 8);
    }
}

// ---------------------------------------------------------------------------
// K5: out = x + BN1(Wz * xb_pos) + BN2(Wz2 * xb_neg).  M=256, N=4096, K=128.
// ---------------------------------------------------------------------------
__global__ __launch_bounds__(32) void zproj_kernel(
    const float* __restrict__ x,
    const float* __restrict__ w_z, const float* __restrict__ w_z2,
    const float* __restrict__ g1, const float* __restrict__ b1,
    const float* __restrict__ m1, const float* __restrict__ v1,
    const float* __restrict__ g2, const float* __restrict__ b2,
    const float* __restrict__ m2, const float* __restrict__ v2,
    const _Float16* __restrict__ xbT, float* __restrict__ out) {
    int b = blockIdx.z;
    int o0 = blockIdx.y * 16;
    int q0 = blockIdx.x * 16;
    const _Float16* xbP = xbT + (size_t)(b * 2 + 0) * HW * PL;
    const _Float16* xbN = xbT + (size_t)(b * 2 + 1) * HW * PL;

    v8f accP = {}, accN = {};
#pragma unroll
    for (int kc = 0; kc < 4; ++kc) {
        v16h aP = load_a_f32(w_z, PL, o0, kc * 32);
        v16h aN = load_a_f32(w_z2, PL, o0, kc * 32);
        v16h bP = load_b_t(xbP, PL, kc * 32, q0);
        v16h bN = load_b_t(xbN, PL, kc * 32, q0);
        accP = __builtin_amdgcn_wmma_f32_16x16x32_f16(false, aP, false, bP,
                                                      (short)0, accP, false, false);
        accN = __builtin_amdgcn_wmma_f32_16x16x32_f16(false, aN, false, bN,
                                                      (short)0, accN, false, false);
    }

    int lane = threadIdx.x & 31, mr = lane & 15, half = lane >> 4;
#pragma unroll
    for (int v = 0; v < 8; ++v) {
        int o = o0 + v + 8 * half;
        float A1 = g1[o] * rsqrtf(v1[o] + 1e-5f);
        float B1 = b1[o] - m1[o] * A1;
        float A2 = g2[o] * rsqrtf(v2[o] + 1e-5f);
        float B2 = b2[o] - m2[o] * A2;
        size_t idx = ((size_t)(b * CIN + o)) * HW + q0 + mr;
        out[idx] = x[idx] + accP[v] * A1 + B1 + accN[v] * A2 + B2;
    }
}

// ---------------------------------------------------------------------------
// Host launcher
// ---------------------------------------------------------------------------
extern "C" void kernel_launch(void* const* d_in, const int* in_sizes, int n_in,
                              void* d_out, int out_size, void* d_ws, size_t ws_size,
                              hipStream_t stream) {
    (void)in_sizes; (void)n_in; (void)out_size; (void)ws_size;

    const float* x      = (const float*)d_in[0];
    const float* click  = (const float*)d_in[1];
    const float* w_t    = (const float*)d_in[2];
    const float* w_p    = (const float*)d_in[3];
    const float* w_g    = (const float*)d_in[4];
    const float* w_z    = (const float*)d_in[5];
    const float* w_z2   = (const float*)d_in[6];
    const float* bn1_g  = (const float*)d_in[7];
    const float* bn1_b  = (const float*)d_in[8];
    const float* bn1_m  = (const float*)d_in[9];
    const float* bn1_v  = (const float*)d_in[10];
    const float* bn2_g  = (const float*)d_in[11];
    const float* bn2_b  = (const float*)d_in[12];
    const float* bn2_m  = (const float*)d_in[13];
    const float* bn2_v  = (const float*)d_in[14];
    const float* dw_w   = (const float*)d_in[15];
    const float* dw_b   = (const float*)d_in[16];
    const float* pw_w   = (const float*)d_in[17];
    const float* pw_b   = (const float*)d_in[18];
    const float* head_w = (const float*)d_in[19];
    const float* head_b = (const float*)d_in[20];

    float* out = (float*)d_out;
    const size_t OUT_X = (size_t)BATCH * CIN * HW;          // 2,097,152 floats
    float* gate_pos_out = out + OUT_X;                      // [2][4096]
    float* gate_neg_out = out + OUT_X + (size_t)BATCH * HW; // [2][4096]

    // workspace layout (bytes)
    char* ws = (char*)d_ws;
    _Float16* xT  = (_Float16*)(ws + 0);                    // 2*4096*256 f16 = 4 MB
    _Float16* tT  = (_Float16*)(ws + (4u << 20));           // 2*4096*128 f16 = 2 MB
    _Float16* pT  = (_Float16*)(ws + (6u << 20));           // 2 MB
    _Float16* gC  = (_Float16*)(ws + (8u << 20));           // 2 MB
    _Float16* xbT = (_Float16*)(ws + (10u << 20));          // 2*2*4096*128 f16 = 4 MB
    float* wclick = (float*)(ws + (14u << 20));             // 2*2*4096 f32 = 64 KB
    float* gpos_s = (float*)(ws + (14u << 20) + 65536);     // 32 KB
    float* gneg_s = (float*)(ws + (14u << 20) + 65536 + 32768);

    // K0: convert x -> xT (f16, q-major), click weights
    {
        int n = BATCH * HW * CIN;
        cvt_kernel<<<(n + 255) / 256, 256, 0, stream>>>(x, xT, click, wclick);
    }
    // K1: t/p/g projections (WMMA)
    {
        dim3 grid(HW / 16, 24, BATCH);
        proj_kernel<<<grid, 32, 0, stream>>>(w_t, w_p, w_g, xT, tT, pT, gC);
    }
    // K2: gate head
    {
        gate_kernel<<<BATCH * HW, 256, 0, stream>>>(x, dw_w, dw_b, pw_w, pw_b,
                                                    head_w, head_b,
                                                    gate_pos_out, gate_neg_out,
                                                    gpos_s, gneg_s);
    }
    // K4: fused non-local attention, both branches (WMMA + online softmax)
    {
        dim3 grid(HW / 16, 2, BATCH);
        attn_kernel<<<grid, 32, 0, stream>>>(tT, pT, gC, wclick, gpos_s, gneg_s, xbT);
    }
    // K5: z-projection + BN + residual (WMMA)
    {
        dim3 grid(HW / 16, CIN / 16, BATCH);
        zproj_kernel<<<grid, 32, 0, stream>>>(x, w_z, w_z2,
                                              bn1_g, bn1_b, bn1_m, bn1_v,
                                              bn2_g, bn2_b, bn2_m, bn2_v,
                                              xbT, out);
    }
}
